// SubEncoder_14001593385671
// MI455X (gfx1250) — compile-verified, matrix operand order checked by reading, not run on hardware
//
#include <hip/hip_runtime.h>

// ---------------------------------------------------------------------------
// Transformer encoder block for MI455X (gfx1250, wave32, WMMA bf16)
//   B=2, S=2048, D=1024, H=16, DK=64, FF=4096, fp32 I/O
// ---------------------------------------------------------------------------

typedef __attribute__((ext_vector_type(16))) __bf16 v16bf;
typedef __attribute__((ext_vector_type(8)))  float  v8f;

using u16 = unsigned short;
using u32 = unsigned int;

static constexpr int Hh = 16;
static constexpr int Dd = 1024;
static constexpr int DKk = 64;
static constexpr int FFf = 4096;
static constexpr int Bb = 2;
static constexpr int Ss = 2048;
static constexpr int BS = Bb * Ss;       // 4096 rows
static constexpr float LN_EPS = 1e-5f;

// ---------------- helpers ---------------------------------------------------

__device__ __forceinline__ u16 f2bf(float f) {
    u32 u = __float_as_uint(f);
    u32 r = u + 0x7FFFu + ((u >> 16) & 1u);   // round-to-nearest-even
    return (u16)(r >> 16);
}

union Op32 { uint4 q[2]; v16bf v; };

// Load one WMMA 16x16x32 bf16 operand (A layout; B uses the same pattern on
// the transposed [N,K] matrix).  `base` points at a row-major bf16 tile in
// LDS with `stride` elements per row, `kbase` selects the 32-wide K chunk.
// Lanes 0-15: K = kbase+{0..7,16..23};  lanes 16-31: K = kbase+{8..15,24..31}.
__device__ __forceinline__ v16bf lds_load_op(const u16* base, int row, int stride,
                                             int kbase, int lane) {
    const u16* p = base + row * stride + kbase + ((lane >> 4) & 1) * 8;
    Op32 u;
    u.q[0] = *reinterpret_cast<const uint4*>(p);
    u.q[1] = *reinterpret_cast<const uint4*>(p + 16);
    return u.v;
}

__device__ __forceinline__ v8f wmma_bf16(v16bf a, v16bf b, v8f c) {
    return __builtin_amdgcn_wmma_f32_16x16x32_bf16(false, a, false, b,
                                                   (short)0, c, false, false);
}

// Cross-lane XOR reduction step within 16-lane halves of the wave32.
// Prefer v_permlane16_b32 (VALU, co-executes with WMMA) over ds_bpermute.
template <int MASK>
__device__ __forceinline__ float xor16f(float v) {
#if __has_builtin(__builtin_amdgcn_permlane16)
    u32 s0, s1;
    if constexpr (MASK == 1)      { s0 = 0x67452301u; s1 = 0xEFCDAB89u; }
    else if constexpr (MASK == 2) { s0 = 0x54761032u; s1 = 0xDCFE98BAu; }
    else if constexpr (MASK == 4) { s0 = 0x32107654u; s1 = 0xBA98FEDCu; }
    else                          { s0 = 0xFEDCBA98u; s1 = 0x76543210u; }
    u32 u = __float_as_uint(v);
    u32 r = __builtin_amdgcn_permlane16(u, u, s0, s1, false, false);
    return __uint_as_float(r);
#else
    return __shfl_xor(v, MASK);
#endif
}

// ---------------- fp32 -> bf16 conversion -----------------------------------

__global__ __launch_bounds__(256) void cvt_bf16_kernel(const float* __restrict__ x,
                                                       u16* __restrict__ y, int n) {
    for (int i = blockIdx.x * blockDim.x + threadIdx.x; i < n;
         i += gridDim.x * blockDim.x)
        y[i] = f2bf(x[i]);
}

// ---------------- generic WMMA GEMM -----------------------------------------
// C[M,N] = A[M,K](bf16) * Bt[N,K](bf16)^T + bias[N], with epilogue modes.
// Block tile 128x128, K-step 64, 8 waves; wave tile 32x64 (2x4 WMMA tiles).

enum { MODE_F32 = 0, MODE_RELU_BF16 = 1, MODE_QPROJ = 2, MODE_VPROJ = 3 };

template <int MODE>
__global__ __launch_bounds__(256) void gemm_bf16_kernel(
    const u16* __restrict__ A, const u16* __restrict__ Bt,
    const float* __restrict__ bias, float* __restrict__ Cf,
    u16* __restrict__ Cb, int M, int N, int K) {
    __shared__ u16 As[128 * 64];   // 16 KB
    __shared__ u16 Bs[128 * 64];   // 16 KB

    const int tid  = threadIdx.x;
    const int lane = tid & 31;
    const int ln   = lane & 15;
    const int half = lane >> 4;
    const int wave = tid >> 5;
    const int wr   = (wave & 3) * 32;   // wave row offset within 128
    const int wc   = (wave >> 2) * 64;  // wave col offset within 128
    const int m0   = blockIdx.y * 128;
    const int n0   = blockIdx.x * 128;

    v8f acc[2][4] = {};

    for (int kc = 0; kc < K; kc += 64) {
        __syncthreads();
        // stage A,B tiles: 128 rows x 64 K each = 1024 x 16B chunks, 4+4/thread
#pragma unroll
        for (int i = 0; i < 4; ++i) {
            int chunk = i * 256 + tid;            // coalesced across threads
            int row = chunk >> 3, q = chunk & 7;
            *reinterpret_cast<uint4*>(&As[row * 64 + q * 8]) =
                *reinterpret_cast<const uint4*>(&A[(size_t)(m0 + row) * K + kc + q * 8]);
            *reinterpret_cast<uint4*>(&Bs[row * 64 + q * 8]) =
                *reinterpret_cast<const uint4*>(&Bt[(size_t)(n0 + row) * K + kc + q * 8]);
        }
        if (kc + 64 < K) {   // keep HBM ahead of the LDS stage
            int chunk = tid;
            int row = chunk >> 3, q = chunk & 7;
            __builtin_prefetch(&A[(size_t)(m0 + row) * K + kc + 64 + q * 8], 0, 0);
            __builtin_prefetch(&Bt[(size_t)(n0 + row) * K + kc + 64 + q * 8], 0, 0);
        }
        __syncthreads();

        v16bf a[2][2];
#pragma unroll
        for (int c = 0; c < 2; ++c)
#pragma unroll
            for (int i = 0; i < 2; ++i)
                a[c][i] = lds_load_op(As, wr + i * 16 + ln, 64, c * 32, lane);

#pragma unroll
        for (int c = 0; c < 2; ++c) {
#pragma unroll
            for (int j = 0; j < 4; ++j) {
                v16bf b = lds_load_op(Bs, wc + j * 16 + ln, 64, c * 32, lane);
                acc[0][j] = wmma_bf16(a[c][0], b, acc[0][j]);
                acc[1][j] = wmma_bf16(a[c][1], b, acc[1][j]);
            }
        }
    }

    // Epilogue.  C layout: VGPR r -> row r + 8*half, col = lane&15.
#pragma unroll
    for (int i = 0; i < 2; ++i) {
#pragma unroll
        for (int j = 0; j < 4; ++j) {
            const int col = n0 + wc + j * 16 + ln;
            const float bval = bias[col];
#pragma unroll
            for (int r = 0; r < 8; ++r) {
                const int row = m0 + wr + i * 16 + r + half * 8;
                float val = acc[i][j][r] + bval;
                if constexpr (MODE == MODE_F32) {
                    Cf[(size_t)row * N + col] = val;
                } else if constexpr (MODE == MODE_RELU_BF16) {
                    Cb[(size_t)row * N + col] = f2bf(fmaxf(val, 0.0f));
                } else if constexpr (MODE == MODE_QPROJ) {
                    // row = b*S+s ; col = h*DK+kk  ->  [b,h,s,kk] bf16
                    int h = col >> 6, kk = col & 63;
                    int s = row & (Ss - 1), bidx = row >> 11;
                    Cb[(((size_t)(bidx * Hh + h) * Ss + s) * DKk) + kk] = f2bf(val);
                } else {  // MODE_VPROJ: [b,h,kk,s] (V transposed)
                    int h = col >> 6, kk = col & 63;
                    int s = row & (Ss - 1), bidx = row >> 11;
                    Cb[(((size_t)(bidx * Hh + h) * DKk + kk) * Ss) + s] = f2bf(val);
                }
            }
        }
    }
}

// ---------------- fused flash attention -------------------------------------
// grid = (S/64, B*H), block = 128 (4 waves).  Each wave owns 16 query rows.
// 64-key tiles: 8 score WMMAs + 8 PV WMMAs per barrier pair.
// Q,K: bf16 [B*H, S, DK] ; VT: bf16 [B*H, DK, S] ; O: bf16 [B, S, H*DK].

__global__ __launch_bounds__(128) void flash_attn_kernel(
    const u16* __restrict__ Q, const u16* __restrict__ Km,
    const u16* __restrict__ VT, u16* __restrict__ O) {
    __shared__ u16 qs[64 * 64];          // 64 q rows x DK      (8 KB)
    __shared__ u16 ks[64 * 64];          // 64 keys  x DK       (8 KB)
    __shared__ u16 vts[64 * 64];         // DK rows  x 64 keys  (8 KB)
    __shared__ u16 ps[4 * 16 * 64];      // per-wave P scratch  (8 KB)

    const int tid  = threadIdx.x;
    const int lane = tid & 31;
    const int ln   = lane & 15;
    const int half = lane >> 4;
    const int wave = tid >> 5;
    const int bh   = blockIdx.y;                 // b*H + h
    const int q0   = blockIdx.x * 64;
    const int rb   = wave * 16;

    // stage Q tile
    {
        const u16* qg = Q + ((size_t)bh * Ss + q0) * DKk;
        int row = tid >> 1, cb = (tid & 1) * 32;
#pragma unroll
        for (int q = 0; q < 4; ++q)
            *reinterpret_cast<uint4*>(&qs[row * 64 + cb + q * 8]) =
                *reinterpret_cast<const uint4*>(&qg[(size_t)row * DKk + cb + q * 8]);
    }
    __syncthreads();

    v16bf aq[2];
#pragma unroll
    for (int c = 0; c < 2; ++c)
        aq[c] = lds_load_op(qs, rb + ln, 64, c * 32, lane);

    float m[8], l[8];
    v8f o[4] = {};
#pragma unroll
    for (int r = 0; r < 8; ++r) { m[r] = -INFINITY; l[r] = 0.0f; }

    u16* psw = ps + wave * 16 * 64;

    for (int kb = 0; kb < Ss; kb += 64) {
        __syncthreads();
        {   // stage K tile: 64 keys x DK
            int row = tid >> 1, cb = (tid & 1) * 32;
            const u16* kg = Km + ((size_t)bh * Ss + kb) * DKk;
#pragma unroll
            for (int q = 0; q < 4; ++q)
                *reinterpret_cast<uint4*>(&ks[row * 64 + cb + q * 8]) =
                    *reinterpret_cast<const uint4*>(&kg[(size_t)row * DKk + cb + q * 8]);
        }
        {   // stage V^T tile: DK rows x 64 keys
            int row = tid >> 1, cb = (tid & 1) * 32;
            const u16* vg = VT + (size_t)bh * DKk * Ss + kb;
#pragma unroll
            for (int q = 0; q < 4; ++q)
                *reinterpret_cast<uint4*>(&vts[row * 64 + cb + q * 8]) =
                    *reinterpret_cast<const uint4*>(&vg[(size_t)row * Ss + cb + q * 8]);
        }
        __syncthreads();

        // scores: [16 q rows] x [64 keys] in 4 tiles, accumulate over DK
        v8f s[4] = {};
#pragma unroll
        for (int c = 0; c < 2; ++c)
#pragma unroll
            for (int t = 0; t < 4; ++t) {
                v16bf bk = lds_load_op(ks, t * 16 + ln, 64, c * 32, lane);
                s[t] = wmma_bf16(aq[c], bk, s[t]);
            }
#pragma unroll
        for (int t = 0; t < 4; ++t) s[t] = s[t] * 0.125f;   // 1/sqrt(DK)

        // online softmax (rows live in 16-lane halves of the wave)
        float rmax[8];
#pragma unroll
        for (int r = 0; r < 8; ++r)
            rmax[r] = fmaxf(fmaxf(s[0][r], s[1][r]), fmaxf(s[2][r], s[3][r]));
#pragma unroll
        for (int r = 0; r < 8; ++r) {
            rmax[r] = fmaxf(rmax[r], xor16f<1>(rmax[r]));
            rmax[r] = fmaxf(rmax[r], xor16f<2>(rmax[r]));
            rmax[r] = fmaxf(rmax[r], xor16f<4>(rmax[r]));
            rmax[r] = fmaxf(rmax[r], xor16f<8>(rmax[r]));
        }

        float alpha[8];
#pragma unroll
        for (int r = 0; r < 8; ++r) {
            float mn = fmaxf(m[r], rmax[r]);
            alpha[r] = __expf(m[r] - mn);
            m[r] = mn;
        }
#pragma unroll
        for (int t = 0; t < 4; ++t)
#pragma unroll
            for (int r = 0; r < 8; ++r)
                s[t][r] = __expf(s[t][r] - m[r]);

        float rsum[8];
#pragma unroll
        for (int r = 0; r < 8; ++r)
            rsum[r] = (s[0][r] + s[1][r]) + (s[2][r] + s[3][r]);
#pragma unroll
        for (int r = 0; r < 8; ++r) {
            rsum[r] += xor16f<1>(rsum[r]);
            rsum[r] += xor16f<2>(rsum[r]);
            rsum[r] += xor16f<4>(rsum[r]);
            rsum[r] += xor16f<8>(rsum[r]);
            l[r] = l[r] * alpha[r] + rsum[r];
        }

        // reshape P (C layout -> A layout) via per-wave LDS round trip
#pragma unroll
        for (int t = 0; t < 4; ++t)
#pragma unroll
            for (int r = 0; r < 8; ++r) {
                int rowp = r + half * 8;
                psw[rowp * 64 + t * 16 + ln] = f2bf(s[t][r]);
            }
        asm volatile("s_wait_dscnt 0x0" ::: "memory");  // wave-local LDS RAW
        v16bf ap[2];
#pragma unroll
        for (int c = 0; c < 2; ++c)
            ap[c] = lds_load_op(psw, ln, 64, c * 32, lane);

        // O = O*alpha + P @ V  (two 32-key WMMA K-steps per col group)
#pragma unroll
        for (int n = 0; n < 4; ++n) {
#pragma unroll
            for (int r = 0; r < 8; ++r) o[n][r] *= alpha[r];
#pragma unroll
            for (int c = 0; c < 2; ++c) {
                v16bf bv = lds_load_op(vts, n * 16 + ln, 64, c * 32, lane);
                o[n] = wmma_bf16(ap[c], bv, o[n]);
            }
        }
    }

    // finalize: O /= l, write concatenated-head layout [b, s, h*DK + col]
    const int bidx = bh >> 4, h = bh & 15;
#pragma unroll
    for (int r = 0; r < 8; ++r) {
        float inv = 1.0f / l[r];
        int srow = q0 + rb + r + half * 8;
#pragma unroll
        for (int n = 0; n < 4; ++n) {
            int col = h * DKk + n * 16 + ln;
            O[((size_t)bidx * Ss + srow) * (Hh * DKk) + col] = f2bf(o[n][r] * inv);
        }
    }
}

// ---------------- residual + LayerNorm --------------------------------------
// one row per block; 256 threads * 4 elems = D

template <bool WRITE_BF16>
__global__ __launch_bounds__(256) void add_ln_kernel(
    const float* __restrict__ a, const float* __restrict__ b,
    const float* __restrict__ g, const float* __restrict__ be,
    float* __restrict__ oF, u16* __restrict__ oB) {
    __shared__ float ss[256], sq[256];
    const int row = blockIdx.x, tid = threadIdx.x;
    const int i0 = tid * 4;

    float4 xa = *reinterpret_cast<const float4*>(a + (size_t)row * Dd + i0);
    float4 xb = *reinterpret_cast<const float4*>(b + (size_t)row * Dd + i0);
    float v[4] = { xa.x + xb.x, xa.y + xb.y, xa.z + xb.z, xa.w + xb.w };

    float s = v[0] + v[1] + v[2] + v[3];
    float q = v[0] * v[0] + v[1] * v[1] + v[2] * v[2] + v[3] * v[3];
    ss[tid] = s; sq[tid] = q;
    __syncthreads();
    for (int st = 128; st > 0; st >>= 1) {
        if (tid < st) { ss[tid] += ss[tid + st]; sq[tid] += sq[tid + st]; }
        __syncthreads();
    }
    const float mu   = ss[0] * (1.0f / Dd);
    const float var  = sq[0] * (1.0f / Dd) - mu * mu;
    const float rstd = rsqrtf(var + LN_EPS);

    float4 gg = *reinterpret_cast<const float4*>(g + i0);
    float4 bb = *reinterpret_cast<const float4*>(be + i0);
    float o0 = (v[0] - mu) * rstd * gg.x + bb.x;
    float o1 = (v[1] - mu) * rstd * gg.y + bb.y;
    float o2 = (v[2] - mu) * rstd * gg.z + bb.z;
    float o3 = (v[3] - mu) * rstd * gg.w + bb.w;

    *reinterpret_cast<float4*>(oF + (size_t)row * Dd + i0) = make_float4(o0, o1, o2, o3);
    if constexpr (WRITE_BF16) {
        u16* p = oB + (size_t)row * Dd + i0;
        p[0] = f2bf(o0); p[1] = f2bf(o1); p[2] = f2bf(o2); p[3] = f2bf(o3);
    }
}

// ---------------- launcher ---------------------------------------------------

extern "C" void kernel_launch(void* const* d_in, const int* in_sizes, int n_in,
                              void* d_out, int out_size, void* d_ws, size_t ws_size,
                              hipStream_t stream) {
    const float* x  = (const float*)d_in[0];
    const float* wq = (const float*)d_in[1];  const float* bq = (const float*)d_in[2];
    const float* wk = (const float*)d_in[3];  const float* bk = (const float*)d_in[4];
    const float* wv = (const float*)d_in[5];  const float* bv = (const float*)d_in[6];
    const float* wo = (const float*)d_in[7];  const float* bo = (const float*)d_in[8];
    const float* g1 = (const float*)d_in[9];  const float* be1 = (const float*)d_in[10];
    const float* w1 = (const float*)d_in[11]; const float* b1 = (const float*)d_in[12];
    const float* w2 = (const float*)d_in[13]; const float* b2 = (const float*)d_in[14];
    const float* g2 = (const float*)d_in[15]; const float* be2 = (const float*)d_in[16];
    float* out = (float*)d_out;

    char* ws = (char*)d_ws;
    const size_t MB = 1024 * 1024;
    u16* XB  = (u16*)(ws + 0);          //  8 MB  x bf16 [BS, D]
    u16* WQB = (u16*)(ws + 8  * MB);    //  2 MB
    u16* WKB = (u16*)(ws + 10 * MB);    //  2 MB
    u16* WVB = (u16*)(ws + 12 * MB);    //  2 MB
    u16* WOB = (u16*)(ws + 14 * MB);    //  2 MB
    u16* W1B = (u16*)(ws + 16 * MB);    //  8 MB
    u16* W2B = (u16*)(ws + 24 * MB);    //  8 MB
    u16* Qb  = (u16*)(ws + 32 * MB);    //  8 MB  [B,H,S,DK]
    u16* Kb  = (u16*)(ws + 40 * MB);    //  8 MB  [B,H,S,DK]
    u16* VTb = (u16*)(ws + 48 * MB);    //  8 MB  [B,H,DK,S]
    u16* Ob  = (u16*)(ws + 56 * MB);    //  8 MB  [B,S,H*DK]
    float* ATT = (float*)(ws + 64 * MB);   // 16 MB [BS, D]
    float* N1F = (float*)(ws + 80 * MB);   // 16 MB
    u16*   N1B = (u16*)(ws + 96 * MB);     //  8 MB
    u16*   FF1 = (u16*)(ws + 104 * MB);    // 32 MB [BS, FF]
    float* FF2 = (float*)(ws + 136 * MB);  // 16 MB

    // 1. bf16 conversions
    auto cvt = [&](const float* src, u16* dst, int n) {
        int blocks = min((n + 255) / 256, 4096);
        cvt_bf16_kernel<<<blocks, 256, 0, stream>>>(src, dst, n);
    };
    cvt(x,  XB,  BS * Dd);
    cvt(wq, WQB, Hh * DKk * Dd);
    cvt(wk, WKB, Hh * DKk * Dd);
    cvt(wv, WVB, Hh * DKk * Dd);
    cvt(wo, WOB, Dd * Dd);
    cvt(w1, W1B, FFf * Dd);
    cvt(w2, W2B, Dd * FFf);

    // 2. QKV projections (N = H*DK = 1024, K = 1024)
    dim3 gP(Dd / 128, BS / 128);
    gemm_bf16_kernel<MODE_QPROJ><<<gP, 256, 0, stream>>>(XB, WQB, bq, nullptr, Qb,  BS, Dd, Dd);
    gemm_bf16_kernel<MODE_QPROJ><<<gP, 256, 0, stream>>>(XB, WKB, bk, nullptr, Kb,  BS, Dd, Dd);
    gemm_bf16_kernel<MODE_VPROJ><<<gP, 256, 0, stream>>>(XB, WVB, bv, nullptr, VTb, BS, Dd, Dd);

    // 3. fused attention
    flash_attn_kernel<<<dim3(Ss / 64, Bb * Hh), 128, 0, stream>>>(Qb, Kb, VTb, Ob);

    // 4. output projection
    gemm_bf16_kernel<MODE_F32><<<gP, 256, 0, stream>>>(Ob, WOB, bo, ATT, nullptr, BS, Dd, Dd);

    // 5. LN1 (residual with x), keep bf16 mirror for FFN input
    add_ln_kernel<true><<<BS, 256, 0, stream>>>(ATT, x, g1, be1, N1F, N1B);

    // 6. FFN
    gemm_bf16_kernel<MODE_RELU_BF16><<<dim3(FFf / 128, BS / 128), 256, 0, stream>>>(
        N1B, W1B, b1, nullptr, FF1, BS, FFf, Dd);
    gemm_bf16_kernel<MODE_F32><<<gP, 256, 0, stream>>>(FF1, W2B, b2, FF2, nullptr, BS, Dd, FFf);

    // 7. LN2 -> output
    add_ln_kernel<false><<<BS, 256, 0, stream>>>(FF2, N1F, g2, be2, out, nullptr);
}
